// FusedTriangleMultiplication_51178830299628
// MI455X (gfx1250) — matrix operand order repaired and checked by reading, not compiled, not run on hardware
//
#include <hip/hip_runtime.h>
#include <hip/hip_bf16.h>
#include <stdint.h>

// ---------------------------------------------------------------------------
// Fused AlphaFold triangle-multiplication (outgoing) for MI455X / gfx1250.
// All GEMMs on v_wmma_f32_16x16x32_bf16 (wave32). The dominant einsum kernel
// streams its K-slabs with GLOBAL_LOAD_ASYNC_TO_LDS_B128 (ASYNCcnt) into
// double-buffered LDS so the matrix pipes never wait on the VGPR staging path.
// ---------------------------------------------------------------------------

#define N_RES 768
#define C_Z   128
#define C_HID 128
#define NPOS  (N_RES * N_RES)          // 589824 flattened (i,k) / (i,j) positions

typedef __bf16 bf16;
typedef bf16  v16bf __attribute__((ext_vector_type(16)));
typedef bf16  v8bf  __attribute__((ext_vector_type(8)));
typedef float v8f   __attribute__((ext_vector_type(8)));

#define V8F_ZERO {0.f,0.f,0.f,0.f,0.f,0.f,0.f,0.f}

__device__ __forceinline__ bf16 to_bf16(float f) {
  uint32_t u = __builtin_bit_cast(uint32_t, f);
  u += 0x7FFFu + ((u >> 16) & 1u);               // round-to-nearest-even
  uint16_t h = (uint16_t)(u >> 16);
  return __builtin_bit_cast(bf16, h);
}

__device__ __forceinline__ v16bf frag16(const bf16* p0, const bf16* p1) {
  v8bf lo = *(const v8bf*)p0;
  v8bf hi = *(const v8bf*)p1;
  return __builtin_shufflevector(lo, hi, 0,1,2,3,4,5,6,7,8,9,10,11,12,13,14,15);
}

// A fragment: 16(M)x32(K) bf16, source row-major (rows=M, leading dim ld).
// ISA layout: lanes 0-15 -> M=lane, K {0..7,16..23}; lanes 16-31 -> K {8..15,24..31}.
__device__ __forceinline__ v16bf load_frag_A(const bf16* p, int ld) {
  int lane = threadIdx.x & 31;
  int hl = lane >> 4, r = lane & 15;
  const bf16* b = p + r * ld + hl * 8;
  return frag16(b, b + 16);
}

// B fragment: 32(K)x16(N); source stores B^T row-major (rows=N, leading dim ld).
// Lane n<16 holds column N=n with K {0..7,16..23}; lanes 16-31 hold K {8..15,24..31}.
__device__ __forceinline__ v16bf load_frag_B(const bf16* p, int ld) {
  int lane = threadIdx.x & 31;
  int hl = lane >> 4, r = lane & 15;
  const bf16* b = p + r * ld + hl * 8;
  return frag16(b, b + 16);
}

__device__ __forceinline__ v8f wmma_bf16(v16bf a, v16bf b, v8f c) {
  return __builtin_amdgcn_wmma_f32_16x16x32_bf16(
      /*neg_a=*/false, a, /*neg_b=*/false, b,
      /*c_mod=*/(short)0, c, /*reuse_a=*/false, /*reuse_b=*/false);
}

__device__ __forceinline__ float sigmoidf(float x) {
  return 1.0f / (1.0f + __expf(-x));
}

// ---- CDNA5 async global->LDS copy (ASYNCcnt path), 16B per lane -----------
// the LDS address comes from the vdst VGPR; low 32 bits of a generic LDS pointer are the
// wave-relative LDS byte offset (flat aperture truncation, ISA 10.2).
__device__ __forceinline__ void async_b128(const bf16* lds_dst, const bf16* gsrc) {
  uint32_t lds_off = (uint32_t)(uintptr_t)lds_dst;
  asm volatile("global_load_async_to_lds_b128 %0, %1, off"
               :: "v"(lds_off), "v"(gsrc) : "memory");
}
__device__ __forceinline__ void wait_asynccnt0() {
  asm volatile("s_wait_asynccnt 0x0" ::: "memory");
}

// ---------------------------------------------------------------------------
// Kernel 0: cast the four weight matrices to bf16 (L2-resident, read many times)
// dst layout: [0,32768) proj_w  [32768,65536) gate_w  [65536,81920) out_w
//             [81920,98304) gl_w
// ---------------------------------------------------------------------------
__global__ __launch_bounds__(256)
void k_convert(const float* __restrict__ pw, const float* __restrict__ gw,
               const float* __restrict__ ow, const float* __restrict__ gl,
               bf16* __restrict__ dst) {
  int i = blockIdx.x * 256 + threadIdx.x;
  if (i < 32768)        dst[i] = to_bf16(pw[i]);
  else if (i < 65536)   dst[i] = to_bf16(gw[i - 32768]);
  else if (i < 81920)   dst[i] = to_bf16(ow[i - 65536]);
  else if (i < 98304)   dst[i] = to_bf16(gl[i - 81920]);
}

// ---------------------------------------------------------------------------
// Kernel 1: LayerNorm + proj(128->256) + gate(128->256) + sigmoid + mask.
// One block = 16 flattened (i,k) positions, 256 threads (8 waves).
// Wave w computes outputs [w*32, w*32+32) for both proj and gate via WMMA.
// Results written channel-major: Lbuf[c][pos], Rbuf[c][pos] (bf16) so the
// einsum kernel gets K-contiguous tiles.
// ---------------------------------------------------------------------------
__global__ __launch_bounds__(256)
void k_ln_proj_gate(const float* __restrict__ act, const float* __restrict__ mask,
                    const float* __restrict__ lnw, const float* __restrict__ lnb,
                    const bf16* __restrict__ projW, const float* __restrict__ projB,
                    const bf16* __restrict__ gateW, const float* __restrict__ gateB,
                    bf16* __restrict__ Lbuf, bf16* __restrict__ Rbuf) {
  __shared__ __align__(16) bf16 xbf[16 * 128];   // LN'd activations (A matrix)
  __shared__ __align__(16) bf16 ybf[256 * 16];   // gated proj, [out_chan][pos]
  const int tid = threadIdx.x;
  const int P0  = blockIdx.x * 16;

  { // ---- LayerNorm: 16 threads per position, 8 channels each, shuffle-reduce
    int pos = tid >> 4;
    int cg  = (tid & 15) * 8;
    const float* src = act + (size_t)(P0 + pos) * C_Z + cg;
    float4 a0 = ((const float4*)src)[0];
    float4 a1 = ((const float4*)src)[1];
    float x[8] = {a0.x, a0.y, a0.z, a0.w, a1.x, a1.y, a1.z, a1.w};
    float s1 = 0.f, s2 = 0.f;
    #pragma unroll
    for (int e = 0; e < 8; e++) { s1 += x[e]; s2 += x[e] * x[e]; }
    #pragma unroll
    for (int m = 1; m < 16; m <<= 1) {
      s1 += __shfl_xor(s1, m, 16);
      s2 += __shfl_xor(s2, m, 16);
    }
    float mean = s1 * (1.f / 128.f);
    float var  = s2 * (1.f / 128.f) - mean * mean;
    float rst  = rsqrtf(var + 1e-5f);
    #pragma unroll
    for (int e = 0; e < 8; e++) {
      float xn = (x[e] - mean) * rst * lnw[cg + e] + lnb[cg + e];
      xbf[pos * 128 + cg + e] = to_bf16(xn);
    }
  }
  __syncthreads();

  { // ---- WMMA proj + gate, fused sigmoid gating
    const int w = tid >> 5, lane = tid & 31, hl = lane >> 4, r = lane & 15;
    const int n0 = w * 32;
    v8f accP0 = V8F_ZERO, accP1 = V8F_ZERO, accG0 = V8F_ZERO, accG1 = V8F_ZERO;
    #pragma unroll
    for (int kb = 0; kb < 128; kb += 32) {
      v16bf a   = load_frag_A(xbf + kb, 128);
      v16bf bp0 = load_frag_B(projW + (size_t)n0 * 128 + kb, 128);
      v16bf bp1 = load_frag_B(projW + (size_t)(n0 + 16) * 128 + kb, 128);
      v16bf bg0 = load_frag_B(gateW + (size_t)n0 * 128 + kb, 128);
      v16bf bg1 = load_frag_B(gateW + (size_t)(n0 + 16) * 128 + kb, 128);
      accP0 = wmma_bf16(a, bp0, accP0);
      accP1 = wmma_bf16(a, bp1, accP1);
      accG0 = wmma_bf16(a, bg0, accG0);
      accG1 = wmma_bf16(a, bg1, accG1);
    }
    float pb0 = projB[n0 + r], pb1 = projB[n0 + 16 + r];
    float gb0 = gateB[n0 + r], gb1 = gateB[n0 + 16 + r];
    #pragma unroll
    for (int v = 0; v < 8; v++) {
      int row = v + hl * 8;                       // D layout: VGPR v -> row v/v+8
      float mk = mask[P0 + row];
      float y0 = mk * (accP0[v] + pb0) * sigmoidf(accG0[v] + gb0);
      float y1 = mk * (accP1[v] + pb1) * sigmoidf(accG1[v] + gb1);
      ybf[(n0 + r) * 16 + row]      = to_bf16(y0);
      ybf[(n0 + 16 + r) * 16 + row] = to_bf16(y1);
    }
  }
  __syncthreads();

  { // ---- channel-major writeout: each thread streams one output channel row
    int o = tid;
    const uint4* s = (const uint4*)(ybf + o * 16);
    uint4 q0 = s[0], q1 = s[1];
    bf16* dst = (o < C_HID) ? (Lbuf + (size_t)o * NPOS + P0)
                            : (Rbuf + (size_t)(o - C_HID) * NPOS + P0);
    ((uint4*)dst)[0] = q0;
    ((uint4*)dst)[1] = q1;
  }
}

// ---------------------------------------------------------------------------
// Kernel 2: the einsum ikc,jkc->ijc.  Per channel c: O = L * R^T (768x768x768).
// Block = one 128x128 (i,j) tile of one channel. 8 waves in a 4x2 grid, each
// owning 32x64 (8 f32 accumulators). K-slabs (128x32 bf16, ld=40 padding ->
// conflict-free 16B fragment reads) stream via GLOBAL_LOAD_ASYNC_TO_LDS_B128
// into double-buffered LDS; each wave gates the swap on s_wait_asynccnt 0 +
// the workgroup barrier, so WMMA on slab n overlaps the DMA of slab n+1.
// ---------------------------------------------------------------------------
#define LDK 40
__global__ __launch_bounds__(256)
void k_einsum(const bf16* __restrict__ Lbuf, const bf16* __restrict__ Rbuf,
              float* __restrict__ Obuf) {
  __shared__ __align__(16) bf16 Lt[2][128 * LDK];
  __shared__ __align__(16) bf16 Rt[2][128 * LDK];
  const int tid = threadIdx.x;
  const int c   = blockIdx.y;
  const int i0  = (blockIdx.x / 6) * 128;
  const int j0  = (blockIdx.x % 6) * 128;
  const bf16* Lp = Lbuf + (size_t)c * NPOS;
  const bf16* Rp = Rbuf + (size_t)c * NPOS;
  float*      Op = Obuf + (size_t)c * NPOS;

  const int w = tid >> 5, lane = tid & 31, hl = lane >> 4, r = lane & 15;
  const int wi = w & 3, wj = w >> 2;             // 4x2 wave grid; 32(i)x64(j) each

  v8f z = V8F_ZERO;
  v8f acc[2][4];
  #pragma unroll
  for (int ti = 0; ti < 2; ti++)
    #pragma unroll
    for (int tj = 0; tj < 4; tj++)
      acc[ti][tj] = z;

  const int lrow = tid >> 1;                     // 0..127
  const int lch  = (tid & 1) * 16;               // 0 or 16 (bf16 elements)
  const bf16* glBase = Lp + (size_t)(i0 + lrow) * N_RES + lch;
  const bf16* grBase = Rp + (size_t)(j0 + lrow) * N_RES + lch;

  // stage k-slab kb into LDS buffer p: 4 async 16B copies per thread (8KB+8KB)
  auto issue = [&](int p, int kb) {
    const bf16* gl = glBase + kb;
    const bf16* gr = grBase + kb;
    bf16* ldl = &Lt[p][lrow * LDK + lch];
    bf16* ldr = &Rt[p][lrow * LDK + lch];
    async_b128(ldl,     gl);
    async_b128(ldl + 8, gl + 8);
    async_b128(ldr,     gr);
    async_b128(ldr + 8, gr + 8);
  };

  issue(0, 0);                 // prologue fill of buffer 0
  wait_asynccnt0();
  __syncthreads();

  int p = 0;
  for (int kb = 0; kb < N_RES; kb += 32) {
    if (kb + 32 < N_RES) issue(p ^ 1, kb + 32);  // overlap DMA with WMMA

    v16bf af[2], bmat[4];
    #pragma unroll
    for (int ti = 0; ti < 2; ti++)
      af[ti] = load_frag_A(&Lt[p][(wi * 32 + ti * 16) * LDK], LDK);
    #pragma unroll
    for (int tj = 0; tj < 4; tj++)
      bmat[tj] = load_frag_B(&Rt[p][(wj * 64 + tj * 16) * LDK], LDK);
    #pragma unroll
    for (int ti = 0; ti < 2; ti++)
      #pragma unroll
      for (int tj = 0; tj < 4; tj++)
        acc[ti][tj] = wmma_bf16(af[ti], bmat[tj], acc[ti][tj]);

    wait_asynccnt0();          // own DMA for p^1 done before the swap barrier
    __syncthreads();
    p ^= 1;
  }

  #pragma unroll
  for (int ti = 0; ti < 2; ti++) {
    #pragma unroll
    for (int tj = 0; tj < 4; tj++) {
      int rb = i0 + wi * 32 + ti * 16 + hl * 8;
      int cb = j0 + wj * 64 + tj * 16 + r;
      float* dst = Op + (size_t)rb * N_RES + cb;
      #pragma unroll
      for (int v = 0; v < 8; v++) dst[(size_t)v * N_RES] = acc[ti][tj][v];
    }
  }
}

// ---------------------------------------------------------------------------
// Kernel 3: LN over channels + out-linear(128->128) + gate-linear + sigmoid.
// One block = 16 flattened (i,j) positions. Gathers channel-major Obuf into
// position-major LDS, LN via shuffles, two chained WMMA GEMMs (second one
// consumes the first through an LDS bf16 round-trip), fused sigmoid multiply.
// ---------------------------------------------------------------------------
__global__ __launch_bounds__(256)
void k_out(const float* __restrict__ Obuf,
           const float* __restrict__ cnw, const float* __restrict__ cnb,
           const bf16* __restrict__ outW, const float* __restrict__ outB,
           const bf16* __restrict__ glW,  const float* __restrict__ glB,
           float* __restrict__ out) {
  __shared__ __align__(16) float oL[16 * 128];
  __shared__ __align__(16) bf16  nbf[16 * 128];
  __shared__ __align__(16) float yf[16 * 128];
  __shared__ __align__(16) bf16  ybf2[16 * 128];
  const int tid = threadIdx.x;
  const int P0  = blockIdx.x * 16;

  { // gather O[c][pos] -> oL[pos][c]  (coalesced 32B global reads per thread)
    int c = tid >> 1, ph = (tid & 1) * 8;
    const float* src = Obuf + (size_t)c * NPOS + P0 + ph;
    float4 f0 = ((const float4*)src)[0];
    float4 f1 = ((const float4*)src)[1];
    float v[8] = {f0.x, f0.y, f0.z, f0.w, f1.x, f1.y, f1.z, f1.w};
    #pragma unroll
    for (int e = 0; e < 8; e++) oL[(ph + e) * 128 + c] = v[e];
  }
  __syncthreads();

  { // LayerNorm over the 128 channels of each position
    int pos = tid >> 4, cg = (tid & 15) * 8;
    float x[8];
    #pragma unroll
    for (int e = 0; e < 8; e++) x[e] = oL[pos * 128 + cg + e];
    float s1 = 0.f, s2 = 0.f;
    #pragma unroll
    for (int e = 0; e < 8; e++) { s1 += x[e]; s2 += x[e] * x[e]; }
    #pragma unroll
    for (int m = 1; m < 16; m <<= 1) {
      s1 += __shfl_xor(s1, m, 16);
      s2 += __shfl_xor(s2, m, 16);
    }
    float mean = s1 * (1.f / 128.f);
    float var  = s2 * (1.f / 128.f) - mean * mean;
    float rst  = rsqrtf(var + 1e-5f);
    #pragma unroll
    for (int e = 0; e < 8; e++) {
      float xn = (x[e] - mean) * rst * cnw[cg + e] + cnb[cg + e];
      nbf[pos * 128 + cg + e] = to_bf16(xn);
    }
  }
  __syncthreads();

  const int w = tid >> 5, lane = tid & 31, hl = lane >> 4, r = lane & 15;
  const int n0 = w * 16;
  v8f z = V8F_ZERO;

  { // y = LN(O) @ out_w^T + out_b
    v8f acc = z;
    #pragma unroll
    for (int kb = 0; kb < 128; kb += 32)
      acc = wmma_bf16(load_frag_A(nbf + kb, 128),
                      load_frag_B(outW + (size_t)n0 * 128 + kb, 128), acc);
    float ob = outB[n0 + r];
    #pragma unroll
    for (int v = 0; v < 8; v++) {
      int row = v + hl * 8;
      float y = acc[v] + ob;
      yf[row * 128 + n0 + r]   = y;
      ybf2[row * 128 + n0 + r] = to_bf16(y);
    }
  }
  __syncthreads();

  { // gate = sigmoid(y @ gl_w^T + gl_b); out = gate * y
    v8f acc = z;
    #pragma unroll
    for (int kb = 0; kb < 128; kb += 32)
      acc = wmma_bf16(load_frag_A(ybf2 + kb, 128),
                      load_frag_B(glW + (size_t)n0 * 128 + kb, 128), acc);
    float gb = glB[n0 + r];
    #pragma unroll
    for (int v = 0; v < 8; v++) {
      int row = v + hl * 8;
      float g = sigmoidf(acc[v] + gb);
      out[(size_t)(P0 + row) * C_Z + n0 + r] = g * yf[row * 128 + n0 + r];
    }
  }
}

// ---------------------------------------------------------------------------
// Launcher. Workspace layout (bytes):
//   [0,             150994944)  Lbuf  bf16 [128][589824]
//   [150994944,     301989888)  Rbuf  bf16 [128][589824]
//   [301989888,     603979776)  Obuf  f32  [128][589824]
//   [603979776,     +196608)    bf16 weights (proj, gate, out, gl)
// ---------------------------------------------------------------------------
extern "C" void kernel_launch(void* const* d_in, const int* in_sizes, int n_in,
                              void* d_out, int out_size, void* d_ws, size_t ws_size,
                              hipStream_t stream) {
  const float* act   = (const float*)d_in[0];
  const float* mask  = (const float*)d_in[1];
  const float* lnw   = (const float*)d_in[2];
  const float* lnb   = (const float*)d_in[3];
  const float* projW = (const float*)d_in[4];
  const float* projB = (const float*)d_in[5];
  const float* gateW = (const float*)d_in[6];
  const float* gateB = (const float*)d_in[7];
  const float* cnw   = (const float*)d_in[8];
  const float* cnb   = (const float*)d_in[9];
  const float* outW  = (const float*)d_in[10];
  const float* outB  = (const float*)d_in[11];
  const float* glW   = (const float*)d_in[12];
  const float* glB   = (const float*)d_in[13];
  float* out = (float*)d_out;

  char* ws = (char*)d_ws;
  const size_t LBYTES = (size_t)C_HID * NPOS * sizeof(bf16);   // 150,994,944
  const size_t OBYTES = (size_t)C_HID * NPOS * sizeof(float);  // 301,989,888
  bf16*  Lbuf = (bf16*)ws;
  bf16*  Rbuf = (bf16*)(ws + LBYTES);
  float* Obuf = (float*)(ws + 2 * LBYTES);
  bf16*  Wbf  = (bf16*)(ws + 2 * LBYTES + OBYTES);
  bf16* projWbf = Wbf;
  bf16* gateWbf = Wbf + 32768;
  bf16* outWbf  = Wbf + 65536;
  bf16* glWbf   = Wbf + 81920;

  k_convert<<<384, 256, 0, stream>>>(projW, gateW, outW, glW, Wbf);
  k_ln_proj_gate<<<NPOS / 16, 256, 0, stream>>>(act, mask, lnw, lnb,
                                                projWbf, projB, gateWbf, gateB,
                                                Lbuf, Rbuf);
  dim3 g2(36, 128);
  k_einsum<<<g2, 256, 0, stream>>>(Lbuf, Rbuf, Obuf);
  k_out<<<NPOS / 16, 256, 0, stream>>>(Obuf, cnw, cnb, outWbf, outB,
                                       glWbf, glB, out);
  (void)in_sizes; (void)n_in; (void)out_size; (void)ws_size;
}
